// TKANCellPyTorch_77704548319909
// MI455X (gfx1250) — compile-verified, matrix operand order checked by reading, not run on hardware
//
#include <hip/hip_runtime.h>
#include <hip/hip_bf16.h>

// ---------------------------------------------------------------------------
// TKAN cell for MI455X (gfx1250, wave32, WMMA).
// All GEMMs run through v_wmma_f32_16x16x32_bf16 (bf16 in, f32 accumulate).
// Wave tile 32x64 (2x4 WMMA tiles), workgroup 4 waves -> 64x128.
// ---------------------------------------------------------------------------

#define B_SZ    4096
#define IN_DIM  512
#define UNITS   1024
#define SUB_DIM 512
#define NSUB    3
#define GRID_SIZE 5
#define SPLINE_ORDER 3
#define NCOEF (GRID_SIZE + SPLINE_ORDER)   // 8
#define GRID_LO (-3.0f)
#define GRID_HI (3.0f)

typedef __bf16 bf16_t;
typedef __attribute__((ext_vector_type(16))) __bf16 v16bf;
typedef __attribute__((ext_vector_type(8)))  __bf16 v8bf;
typedef __attribute__((ext_vector_type(8)))  float  v8f;

__device__ __forceinline__ unsigned short f32_to_bf16(float f) {
    union { float f; unsigned int u; } v; v.f = f;
    unsigned int r = v.u + 0x7FFFu + ((v.u >> 16) & 1u);   // round-to-nearest-even
    return (unsigned short)(r >> 16);
}

__device__ __forceinline__ float sigmoidf(float x) {
    return 1.0f / (1.0f + __expf(-x));
}

// ---------------------------------------------------------------------------
// Fragment layout for v_wmma_*_16x16x32_bf16 (wave32):
//   lane l (row/col = l&15), K slots: [ (l>>4)*8 .. +7 ] and [ 16+(l>>4)*8 .. +7 ]
//   -> two contiguous 16-byte loads per fragment per lane.
// D layout: lane l, vgpr j -> (m = j + 8*(l>>4), n = l&15).
// ---------------------------------------------------------------------------
__device__ __forceinline__ v16bf load_frag(const bf16_t* __restrict__ p) {
    v8bf lo = *(const v8bf*)p;          // K slots +0..+7
    v8bf hi = *(const v8bf*)(p + 16);   // K slots +16..+23
    v16bf r;
#pragma unroll
    for (int i = 0; i < 8; ++i) { r[i] = lo[i]; r[i + 8] = hi[i]; }
    return r;
}

// K-loop for one operand pair: acc[2][4] covers a 32x64 output tile.
__device__ __forceinline__ void gemm_loop(const bf16_t* __restrict__ A,
                                          const bf16_t* __restrict__ B,
                                          int K, int m0, int n0,
                                          int lrow, int lhi, v8f acc[2][4])
{
    const bf16_t* ap = A + (long)(m0 + lrow) * K + lhi * 8;
    const bf16_t* bp = B + (long)(n0 + lrow) * K + lhi * 8;
    for (int k = 0; k < K; k += 32) {
        v16bf a0 = load_frag(ap + k);
        v16bf a1 = load_frag(ap + (long)16 * K + k);
        v16bf b0 = load_frag(bp + k);
        v16bf b1 = load_frag(bp + (long)16 * K + k);
        v16bf b2 = load_frag(bp + (long)32 * K + k);
        v16bf b3 = load_frag(bp + (long)48 * K + k);
        acc[0][0] = __builtin_amdgcn_wmma_f32_16x16x32_bf16(false, a0, false, b0, (short)0, acc[0][0], false, false);
        acc[0][1] = __builtin_amdgcn_wmma_f32_16x16x32_bf16(false, a0, false, b1, (short)0, acc[0][1], false, false);
        acc[0][2] = __builtin_amdgcn_wmma_f32_16x16x32_bf16(false, a0, false, b2, (short)0, acc[0][2], false, false);
        acc[0][3] = __builtin_amdgcn_wmma_f32_16x16x32_bf16(false, a0, false, b3, (short)0, acc[0][3], false, false);
        acc[1][0] = __builtin_amdgcn_wmma_f32_16x16x32_bf16(false, a1, false, b0, (short)0, acc[1][0], false, false);
        acc[1][1] = __builtin_amdgcn_wmma_f32_16x16x32_bf16(false, a1, false, b1, (short)0, acc[1][1], false, false);
        acc[1][2] = __builtin_amdgcn_wmma_f32_16x16x32_bf16(false, a1, false, b2, (short)0, acc[1][2], false, false);
        acc[1][3] = __builtin_amdgcn_wmma_f32_16x16x32_bf16(false, a1, false, b3, (short)0, acc[1][3], false, false);
    }
}

// ---------------------------------------------------------------------------
// Generic WMMA GEMM:  C[M][N] (f32) = A1[M][K1] * B1t[N][K1]
//                                   + A2[M][K2] * B2t[N][K2]  (optional)
//                                   + bias[N]                 (optional)
// flags bit0: accumulate into existing C
// flags bit1: epilogue  C = sigmoid(acc + bias) * tanh(cnew)   (h_new path)
// ---------------------------------------------------------------------------
__global__ __launch_bounds__(128)
void gemm_bf16_wmma_kernel(const bf16_t* __restrict__ A1,
                           const bf16_t* __restrict__ B1, int K1,
                           const bf16_t* __restrict__ A2,
                           const bf16_t* __restrict__ B2, int K2,
                           const float* __restrict__ bias,
                           float* __restrict__ C, int N,
                           int flags, const float* __restrict__ cnew)
{
    const int lane = threadIdx.x & 31;
    const int wave = threadIdx.x >> 5;
    const int lrow = lane & 15;
    const int lhi  = lane >> 4;
    const int m0 = blockIdx.y * 64  + (wave >> 1) * 32;
    const int n0 = blockIdx.x * 128 + (wave & 1) * 64;

    v8f acc[2][4];
#pragma unroll
    for (int i = 0; i < 2; ++i)
#pragma unroll
        for (int j = 0; j < 4; ++j)
            acc[i][j] = (v8f){};

    if (flags & 1) {
#pragma unroll
        for (int i = 0; i < 2; ++i)
#pragma unroll
            for (int j = 0; j < 4; ++j)
#pragma unroll
                for (int r = 0; r < 8; ++r)
                    acc[i][j][r] =
                        C[(long)(m0 + 16 * i + r + 8 * lhi) * N + n0 + 16 * j + lrow];
    }

    gemm_loop(A1, B1, K1, m0, n0, lrow, lhi, acc);
    if (K2 > 0)
        gemm_loop(A2, B2, K2, m0, n0, lrow, lhi, acc);

    float bs[4];
#pragma unroll
    for (int j = 0; j < 4; ++j)
        bs[j] = bias ? bias[n0 + 16 * j + lrow] : 0.0f;

#pragma unroll
    for (int i = 0; i < 2; ++i) {
#pragma unroll
        for (int j = 0; j < 4; ++j) {
#pragma unroll
            for (int r = 0; r < 8; ++r) {
                long idx = (long)(m0 + 16 * i + r + 8 * lhi) * N + n0 + 16 * j + lrow;
                float v = acc[i][j][r] + bs[j];
                if (flags & 2)
                    v = sigmoidf(v) * tanhf(cnew[idx]);
                C[idx] = v;
            }
        }
    }
}

// ---------------------------------------------------------------------------
// Elementwise / data-movement kernels
// ---------------------------------------------------------------------------
__global__ void convert_bf16_kernel(const float* __restrict__ in,
                                    unsigned short* __restrict__ out, long n)
{
    long stride = (long)gridDim.x * blockDim.x;
    for (long i = (long)blockIdx.x * blockDim.x + threadIdx.x; i < n; i += stride)
        out[i] = f32_to_bf16(in[i]);
}

// in: batch x R x C  ->  out: batch x C x R (bf16)
__global__ void transpose_convert_kernel(const float* __restrict__ in,
                                         unsigned short* __restrict__ out,
                                         int R, int C)
{
    const float* ip = in + (long)blockIdx.z * R * C;
    unsigned short* op = out + (long)blockIdx.z * R * C;
    long n = (long)R * C;
    long stride = (long)gridDim.x * blockDim.x;
    for (long i = (long)blockIdx.x * blockDim.x + threadIdx.x; i < n; i += stride) {
        long r = i / C, c = i % C;
        op[c * (long)R + r] = f32_to_bf16(ip[i]);
    }
}

// scaled spline weight: out[idx] = bf16(sw[idx] * scaler[idx/NCOEF])
__global__ void scale_ssw_kernel(const float* __restrict__ sw,
                                 const float* __restrict__ scaler,
                                 unsigned short* __restrict__ out, long n)
{
    long stride = (long)gridDim.x * blockDim.x;
    for (long i = (long)blockIdx.x * blockDim.x + threadIdx.x; i < n; i += stride)
        out[i] = f32_to_bf16(sw[i] * scaler[i / NCOEF]);
}

// c_new = sigmoid(z_f)*c0 + sigmoid(z_i)*tanh(sigmoid(z_c))
__global__ void gates_kernel(const float* __restrict__ z,
                             const float* __restrict__ c0,
                             float* __restrict__ c_new, long n)
{
    long stride = (long)gridDim.x * blockDim.x;
    for (long idx = (long)blockIdx.x * blockDim.x + threadIdx.x; idx < n; idx += stride) {
        long b = idx / UNITS;
        int  u = (int)(idx % UNITS);
        const float* zr = z + b * (3 * UNITS);
        float gi = sigmoidf(zr[u]);
        float gf = sigmoidf(zr[UNITS + u]);
        float gc = sigmoidf(zr[2 * UNITS + u]);
        c_new[idx] = gf * c0[idx] + gi * tanhf(gc);
    }
}

// Cubic B-spline bases on a uniform grid (Cox-de Boor), 8 outputs.
__device__ __forceinline__ void bspline8(float x, float* out) {
    const float h = (GRID_HI - GRID_LO) / GRID_SIZE;   // 1.2
    float t[12];
#pragma unroll
    for (int j = 0; j < 12; ++j) t[j] = GRID_LO + (j - SPLINE_ORDER) * h;
    float b[11];
#pragma unroll
    for (int j = 0; j < 11; ++j) b[j] = (x >= t[j] && x < t[j + 1]) ? 1.0f : 0.0f;
#pragma unroll
    for (int k = 1; k <= SPLINE_ORDER; ++k) {
        float inv = 1.0f / (k * h);
#pragma unroll
        for (int j = 0; j + k < 11; ++j)
            b[j] = (x - t[j]) * inv * b[j] + (t[j + k + 1] - x) * inv * b[j + 1];
    }
#pragma unroll
    for (int j = 0; j < NCOEF; ++j) out[j] = b[j];
}

__global__ void silu_spline_kernel(const float* __restrict__ agg,
                                   unsigned short* __restrict__ silub,
                                   unsigned short* __restrict__ basesb, long n)
{
    long stride = (long)gridDim.x * blockDim.x;
    for (long idx = (long)blockIdx.x * blockDim.x + threadIdx.x; idx < n; idx += stride) {
        float x = agg[idx];
        silub[idx] = f32_to_bf16(x * sigmoidf(x));
        float bsv[NCOEF];
        bspline8(x, bsv);
#pragma unroll
        for (int j = 0; j < NCOEF; ++j)
            basesb[idx * NCOEF + j] = f32_to_bf16(bsv[j]);
    }
}

// new_sub_states[k][b][s] = rk_h[k][s]*sub_out + rk_x[k][s]*sub_states
__global__ void substate_kernel(const float* __restrict__ sub_out,
                                const float* __restrict__ sub_states,
                                const float* __restrict__ srk,
                                float* __restrict__ out, long n)
{
    long stride = (long)gridDim.x * blockDim.x;
    const long per_k = (long)B_SZ * SUB_DIM;
    for (long idx = (long)blockIdx.x * blockDim.x + threadIdx.x; idx < n; idx += stride) {
        int k = (int)(idx / per_k);
        int s = (int)(idx % SUB_DIM);
        float rh = srk[(long)k * 2 * SUB_DIM + s];
        float rx = srk[(long)k * 2 * SUB_DIM + SUB_DIM + s];
        out[idx] = rh * sub_out[idx] + rx * sub_states[idx];
    }
}

// agg_flat bf16: out[b][k*SUB_DIM+s] = sub_out[k][b][s]
__global__ void aggflat_kernel(const float* __restrict__ sub_out,
                               unsigned short* __restrict__ out, long n)
{
    long stride = (long)gridDim.x * blockDim.x;
    for (long idx = (long)blockIdx.x * blockDim.x + threadIdx.x; idx < n; idx += stride) {
        long b = idx / (NSUB * SUB_DIM);
        int  j = (int)(idx % (NSUB * SUB_DIM));
        int  k = j / SUB_DIM;
        int  s = j % SUB_DIM;
        out[idx] = f32_to_bf16(sub_out[((long)k * B_SZ + b) * SUB_DIM + s]);
    }
}

// ---------------------------------------------------------------------------
// Host-side launch
// ---------------------------------------------------------------------------
static inline void launch_gemm(hipStream_t s,
                               const void* A1, const void* B1, int K1,
                               const void* A2, const void* B2, int K2,
                               const float* bias, float* C, int M, int N,
                               int flags, const float* cnew)
{
    dim3 grid(N / 128, M / 64);
    gemm_bf16_wmma_kernel<<<grid, 128, 0, s>>>(
        (const bf16_t*)A1, (const bf16_t*)B1, K1,
        (const bf16_t*)A2, (const bf16_t*)B2, K2,
        bias, C, N, flags, cnew);
}

extern "C" void kernel_launch(void* const* d_in, const int* in_sizes, int n_in,
                              void* d_out, int out_size, void* d_ws, size_t ws_size,
                              hipStream_t stream)
{
    (void)in_sizes; (void)n_in; (void)out_size; (void)ws_size;

    const float* x    = (const float*)d_in[0];   // (B, IN_DIM)
    const float* h0   = (const float*)d_in[1];   // (B, UNITS)
    const float* c0   = (const float*)d_in[2];   // (B, UNITS)
    const float* ss   = (const float*)d_in[3];   // (NSUB, B, SUB_DIM)
    const float* kern = (const float*)d_in[4];   // (IN_DIM, 3*UNITS)
    const float* rk   = (const float*)d_in[5];   // (UNITS, 3*UNITS)
    const float* bias = (const float*)d_in[6];   // (3*UNITS)
    const float* ski  = (const float*)d_in[7];   // (NSUB, IN_DIM, SUB_DIM)
    const float* sks  = (const float*)d_in[8];   // (NSUB, SUB_DIM, SUB_DIM)
    const float* srk  = (const float*)d_in[9];   // (NSUB, 2*SUB_DIM)
    const float* bw   = (const float*)d_in[10];  // (NSUB, SUB_DIM(o), SUB_DIM(s))
    const float* sw   = (const float*)d_in[11];  // (NSUB, S, S, NCOEF)
    const float* ssc  = (const float*)d_in[12];  // (NSUB, S, S)
    const float* aggw = (const float*)d_in[13];  // (NSUB*SUB_DIM, UNITS)
    const float* aggb = (const float*)d_in[14];  // (UNITS)

    float* h_new  = (float*)d_out;                                   // (B, UNITS)
    float* c_new  = (float*)d_out + (long)B_SZ * UNITS;              // (B, UNITS)
    float* new_ss = (float*)d_out + 2L * B_SZ * UNITS;               // (NSUB, B, SUB_DIM)

    // ---- workspace layout (256B aligned) ----
    char* ws = (char*)d_ws;
    size_t off = 0;
    auto alloc = [&](size_t bytes) -> char* {
        char* p = ws + off;
        off += (bytes + 255) & ~(size_t)255;
        return p;
    };
    unsigned short* xb     = (unsigned short*)alloc((size_t)B_SZ * IN_DIM * 2);
    unsigned short* h0b    = (unsigned short*)alloc((size_t)B_SZ * UNITS * 2);
    unsigned short* ssb    = (unsigned short*)alloc((size_t)NSUB * B_SZ * SUB_DIM * 2);
    unsigned short* kTb    = (unsigned short*)alloc((size_t)3 * UNITS * IN_DIM * 2);
    unsigned short* rkTb   = (unsigned short*)alloc((size_t)3 * UNITS * UNITS * 2);
    unsigned short* skiTb  = (unsigned short*)alloc((size_t)NSUB * IN_DIM * SUB_DIM * 2);
    unsigned short* sksTb  = (unsigned short*)alloc((size_t)NSUB * SUB_DIM * SUB_DIM * 2);
    unsigned short* bwb    = (unsigned short*)alloc((size_t)NSUB * SUB_DIM * SUB_DIM * 2);
    unsigned short* sswb   = (unsigned short*)alloc((size_t)NSUB * SUB_DIM * SUB_DIM * NCOEF * 2);
    unsigned short* aggWTb = (unsigned short*)alloc((size_t)UNITS * NSUB * SUB_DIM * 2);
    float*          z      = (float*)alloc((size_t)B_SZ * 3 * UNITS * 4);
    float*          agg    = (float*)alloc((size_t)NSUB * B_SZ * SUB_DIM * 4);
    unsigned short* silub  = (unsigned short*)alloc((size_t)NSUB * B_SZ * SUB_DIM * 2);
    unsigned short* basesb = (unsigned short*)alloc((size_t)NSUB * B_SZ * SUB_DIM * NCOEF * 2);
    float*          subo   = (float*)alloc((size_t)NSUB * B_SZ * SUB_DIM * 4);
    unsigned short* afb    = (unsigned short*)alloc((size_t)B_SZ * NSUB * SUB_DIM * 2);

    const int EB = 2048;   // grid-stride elementwise blocks
    const int ET = 256;

    // 1) bf16 conversions of activations
    convert_bf16_kernel<<<EB, ET, 0, stream>>>(x,  xb,  (long)B_SZ * IN_DIM);
    convert_bf16_kernel<<<EB, ET, 0, stream>>>(h0, h0b, (long)B_SZ * UNITS);
    convert_bf16_kernel<<<EB, ET, 0, stream>>>(ss, ssb, (long)NSUB * B_SZ * SUB_DIM);

    // 2) weights -> bf16 Bt form ([N][K], K contiguous)
    transpose_convert_kernel<<<dim3(EB, 1, 1),    ET, 0, stream>>>(kern, kTb,  IN_DIM,  3 * UNITS);
    transpose_convert_kernel<<<dim3(EB, 1, 1),    ET, 0, stream>>>(rk,   rkTb, UNITS,   3 * UNITS);
    transpose_convert_kernel<<<dim3(EB, 1, NSUB), ET, 0, stream>>>(ski,  skiTb, IN_DIM,  SUB_DIM);
    transpose_convert_kernel<<<dim3(EB, 1, NSUB), ET, 0, stream>>>(sks,  sksTb, SUB_DIM, SUB_DIM);
    transpose_convert_kernel<<<dim3(EB, 1, 1),    ET, 0, stream>>>(aggw, aggWTb, NSUB * SUB_DIM, UNITS);
    convert_bf16_kernel<<<EB, ET, 0, stream>>>(bw, bwb, (long)NSUB * SUB_DIM * SUB_DIM);  // already [o][s]
    scale_ssw_kernel<<<EB, ET, 0, stream>>>(sw, ssc, sswb, (long)NSUB * SUB_DIM * SUB_DIM * NCOEF);

    // 3) gates: z = x@kernel + h0@rk + bias  (WMMA, dual-K)
    launch_gemm(stream, xb, kTb, IN_DIM, h0b, rkTb, UNITS, bias, z,
                B_SZ, 3 * UNITS, 0, nullptr);
    gates_kernel<<<EB, ET, 0, stream>>>(z, c0, c_new, (long)B_SZ * UNITS);

    // 4) agg[k] = x@ski[k] + ss[k]@sks[k]   (WMMA, dual-K)
    for (int k = 0; k < NSUB; ++k) {
        launch_gemm(stream,
                    xb, skiTb + (size_t)k * IN_DIM * SUB_DIM, IN_DIM,
                    ssb + (size_t)k * B_SZ * SUB_DIM,
                    sksTb + (size_t)k * SUB_DIM * SUB_DIM, SUB_DIM,
                    nullptr, agg + (size_t)k * B_SZ * SUB_DIM,
                    B_SZ, SUB_DIM, 0, nullptr);
    }

    // 5) silu(agg) and B-spline bases (bf16)
    silu_spline_kernel<<<EB, ET, 0, stream>>>(agg, silub, basesb,
                                              (long)NSUB * B_SZ * SUB_DIM);

    // 6) sub_out[k] = silu(agg)@bw[k]^T  then  += bases@scaled_sw[k]^T (K=4096)
    for (int k = 0; k < NSUB; ++k) {
        launch_gemm(stream,
                    silub + (size_t)k * B_SZ * SUB_DIM,
                    bwb + (size_t)k * SUB_DIM * SUB_DIM, SUB_DIM,
                    nullptr, nullptr, 0,
                    nullptr, subo + (size_t)k * B_SZ * SUB_DIM,
                    B_SZ, SUB_DIM, 0, nullptr);
        launch_gemm(stream,
                    basesb + (size_t)k * B_SZ * SUB_DIM * NCOEF,
                    sswb + (size_t)k * SUB_DIM * SUB_DIM * NCOEF, SUB_DIM * NCOEF,
                    nullptr, nullptr, 0,
                    nullptr, subo + (size_t)k * B_SZ * SUB_DIM,
                    B_SZ, SUB_DIM, 1 /*accumulate*/, nullptr);
    }

    // 7) new_sub_states
    substate_kernel<<<EB, ET, 0, stream>>>(subo, ss, srk, new_ss,
                                           (long)NSUB * B_SZ * SUB_DIM);

    // 8) agg_flat (B, NSUB*SUB_DIM) bf16
    aggflat_kernel<<<EB, ET, 0, stream>>>(subo, afb, (long)B_SZ * NSUB * SUB_DIM);

    // 9) h_new = sigmoid(agg_flat@aggw + aggb) * tanh(c_new)  (WMMA + fused epilogue)
    launch_gemm(stream, afb, aggWTb, NSUB * SUB_DIM, nullptr, nullptr, 0,
                aggb, h_new, B_SZ, UNITS, 2 /*final epilogue*/, c_new);
}
